// MSTI_28982439313656
// MI455X (gfx1250) — matrix operand order repaired and checked by reading, not compiled, not run on hardware
//
#include <hip/hip_runtime.h>

typedef _Float16 half_t;
typedef __attribute__((ext_vector_type(8)))  _Float16 v8h;
typedef __attribute__((ext_vector_type(16))) _Float16 v16h;
typedef __attribute__((ext_vector_type(8)))  float    v8f;

#define NW 8          // waves per block (wave32)
#define BLOCK 256

struct BlockParams {
  float* x;                                   // in-place activation buffer (d_out)
  const half_t *wq, *wk, *wv, *wd, *w1, *w2;  // f16 weights in workspace
  const float  *bq, *bk, *bv, *bd, *b1, *b2;  // biases (f32, from inputs)
  const float  *g1, *bt1, *g2, *bt2;          // LN gamma/beta
  int S, A, hd, Dq, Dk, Dv, qoff, koff, voff, dim1;
};

__device__ __forceinline__ void lds_fence() {
  __asm__ volatile("s_wait_dscnt 0x0" ::: "memory");
}

__device__ __forceinline__ v8f wmma16(v16h a, v16h b, v8f c) {
  // D = A(16x32 f16) x B(32x16 f16) + C(16x16 f32)
  return __builtin_amdgcn_wmma_f32_16x16x32_f16(false, a, false, b, (short)0, c,
                                                false, false);
}

// ---- fragment loaders --------------------------------------------------
// A-matrix 16x32 f16: lane l holds row m=l&15; element j -> k = j + 16*(l>=16)
__device__ __forceinline__ v16h fragA_lds(const half_t* rb, int stride, int lane) {
  int m = lane & 15, g = lane >> 4;
  const half_t* p = rb + m * stride + (g << 4);
  v8h a0 = *(const v8h*)p;
  v8h a1 = *(const v8h*)(p + 8);
  v16h r;
#pragma unroll
  for (int i = 0; i < 8; i++) { r[i] = a0[i]; r[i + 8] = a1[i]; }
  return r;
}

// masked variant: vector loads (buffers are padded/contiguous), zero k>=kmax in regs
__device__ __forceinline__ v16h fragA_lds_masked(const half_t* rb, int stride,
                                                 int lane, int kmax) {
  int m = lane & 15, g = lane >> 4;
  const half_t* p = rb + m * stride + (g << 4);
  v8h a0 = *(const v8h*)p;
  v8h a1 = *(const v8h*)(p + 8);
  v16h r;
#pragma unroll
  for (int i = 0; i < 8; i++) {
    int k0 = (g << 4) + i;
    int k1 = (g << 4) + 8 + i;
    r[i]     = (k0 < kmax) ? a0[i] : (half_t)0.f;
    r[i + 8] = (k1 < kmax) ? a1[i] : (half_t)0.f;
  }
  return r;
}

// B-matrix 32x16 f16: lane l holds col n=l&15; element j -> k = j + 16*(l>=16)
__device__ __forceinline__ v16h fragB_lds(const half_t* rb, int stride, int lane) {
  int n = lane & 15, g = lane >> 4;
  const half_t* p = rb + n * stride + (g << 4);
  v8h a0 = *(const v8h*)p;
  v8h a1 = *(const v8h*)(p + 8);
  v16h r;
#pragma unroll
  for (int i = 0; i < 8; i++) { r[i] = a0[i]; r[i + 8] = a1[i]; }
  return r;
}

__device__ __forceinline__ v16h fragB_lds_masked(const half_t* rb, int stride,
                                                 int lane, int kmax) {
  int n = lane & 15, g = lane >> 4;
  const half_t* p = rb + n * stride + (g << 4);
  v8h a0 = *(const v8h*)p;
  v8h a1 = *(const v8h*)(p + 8);
  v16h r;
#pragma unroll
  for (int i = 0; i < 8; i++) {
    int k0 = (g << 4) + i;
    int k1 = (g << 4) + 8 + i;
    r[i]     = (k0 < kmax) ? a0[i] : (half_t)0.f;
    r[i + 8] = (k1 < kmax) ? a1[i] : (half_t)0.f;
  }
  return r;
}

// B fragment from global f16 weights laid out [dout][din] row-major
__device__ __forceinline__ v16h fragB_gbl(const half_t* w, int din, int n0,
                                          int k0, int lane) {
  int n = lane & 15, g = lane >> 4;
  const half_t* p = w + (size_t)(n0 + n) * din + k0 + (g << 4);
  v8h a0 = *(const v8h*)p;
  v8h a1 = *(const v8h*)(p + 8);
  v16h r;
#pragma unroll
  for (int i = 0; i < 8; i++) { r[i] = a0[i]; r[i + 8] = a1[i]; }
  return r;
}

// ---- 16-lane-group cross-lane reductions (rows live in half-wave groups) --
__device__ __forceinline__ float gmax16(float v) {
#pragma unroll
  for (int m = 1; m < 16; m <<= 1) v = fmaxf(v, __shfl_xor(v, m, 32));
  return v;
}
__device__ __forceinline__ float gsum16(float v) {
#pragma unroll
  for (int m = 1; m < 16; m <<= 1) v += __shfl_xor(v, m, 32);
  return v;
}

__global__ void cvt_f32_f16(const float* __restrict__ s, half_t* __restrict__ d, int n) {
  int i = blockIdx.x * 256 + threadIdx.x;
  if (i < n) d[i] = (half_t)s[i];
}

// ===========================================================================
// One transformer block, fully fused.  grid = #sequences, block = 256 threads.
// ===========================================================================
__global__ __launch_bounds__(BLOCK) void msti_block_kernel(BlockParams P) {
  extern __shared__ __align__(16) char smem_raw[];
  half_t* sm = (half_t*)smem_raw;

  const int S = P.S, A = P.A, hd = P.hd, Dv = P.Dv;
  const int XS = 136, AS = 136;          // row strides (halves), mult of 8 -> 16B aligned
  const int VtS = S + 8;

  half_t* x_lds = sm;                    // [S][XS]  token channels (all 128), f16
  half_t* QO    = x_lds + S * XS;        // [S][AS]  Q, then attention output O
  half_t* O1    = QO + S * AS;           // [S][AS]  K during attention, out1 after
  half_t* Vt    = O1 + S * AS;           // [A+16][VtS]  V transposed (channel-major)
  half_t* scrA  = Vt + (A + 16) * VtS;   // [NW][16*32]  per-wave transpose scratch

  const int tid  = threadIdx.x;
  const int lane = tid & 31;
  // scalar (SGPR) wave id: makes all work-distribution control flow uniform
  const int wave = __builtin_amdgcn_readfirstlane(tid) >> 5;
  const int g = lane >> 4, ln = lane & 15;
  half_t* scr = scrA + wave * 512;

  // ---- sequence -> global addressing (B=8, T=48, N=256, D=128 hardcoded) ----
  long base0; int tstr;
  if (P.dim1) { int b = blockIdx.x >> 8, n = blockIdx.x & 255;
                base0 = (long)b * 48 * 256 * 128 + (long)n * 128; tstr = 256 * 128; }
  else        { base0 = (long)blockIdx.x * 256 * 128;             tstr = 128; }
  float* xg = P.x;

  // ---- phase 0: zero Vt pad region, load tokens to f16 LDS ----
  for (int i = tid; i < (A + 16) * VtS; i += BLOCK) Vt[i] = (half_t)0.f;
  for (int i = tid; i < S * 32; i += BLOCK) {
    int tok = i >> 5, c4 = (i & 31) << 2;
    const float4 v = *(const float4*)(xg + base0 + (long)tok * tstr + c4);
    half_t* d = x_lds + tok * XS + c4;
    d[0] = (half_t)v.x; d[1] = (half_t)v.y; d[2] = (half_t)v.z; d[3] = (half_t)v.w;
  }
  __syncthreads();

  const int Srt = S >> 4, nct = A >> 4, nvt = Dv >> 4;
  const float qscale = rsqrtf((float)hd);

  // ---- phase 1: Q/K/V GEMMs (Q scaled; K row-major; V transposed) ----
  {
    int total = 3 * Srt * nct;
    for (int it = wave; it < total; it += NW) {
      int mat = it % 3, rem = it / 3, rt = rem % Srt, ct = rem / Srt;
      const half_t* W; const float* bias; int din, off;
      if (mat == 0)      { W = P.wq; bias = P.bq; din = P.Dq; off = P.qoff; }
      else if (mat == 1) { W = P.wk; bias = P.bk; din = P.Dk; off = P.koff; }
      else               { W = P.wv; bias = P.bv; din = P.Dv; off = P.voff; }
      v8f acc = {};
      for (int k0 = 0; k0 < din; k0 += 32) {
        v16h a = fragA_lds(x_lds + rt * 16 * XS + off + k0, XS, lane);
        v16h b = fragB_gbl(W, din, ct * 16, k0, lane);
        acc = wmma16(a, b, acc);
      }
      float bc = bias[ct * 16 + ln];
      // mat is scalar -> three uniform store paths, no per-lane EXEC juggling
      if (mat == 0) {
#pragma unroll
        for (int r = 0; r < 8; r++) {
          int row = rt * 16 + r + (g << 3);
          QO[row * AS + ct * 16 + ln] = (half_t)((acc[r] + bc) * qscale);
        }
      } else if (mat == 1) {
#pragma unroll
        for (int r = 0; r < 8; r++) {
          int row = rt * 16 + r + (g << 3);
          O1[row * AS + ct * 16 + ln] = (half_t)(acc[r] + bc);
        }
      } else {
#pragma unroll
        for (int r = 0; r < 8; r++) {
          int row = rt * 16 + r + (g << 3);
          Vt[(ct * 16 + ln) * VtS + row] = (half_t)(acc[r] + bc);
        }
      }
    }
  }
  __syncthreads();

  // ---- phase 2: flash attention per head / 16-query tile; O overwrites Q ----
  {
    const int nkt = Srt;
    for (int qt = wave; qt < Srt; qt += NW) {
      for (int h = 0; h < 8; h++) {
        int hoff = h * hd;
        v16h aq = fragA_lds_masked(QO + qt * 16 * AS + hoff, AS, lane, hd);
        float om[8], os[8]; v8f oacc = {};
#pragma unroll
        for (int r = 0; r < 8; r++) { om[r] = -1e30f; os[r] = 0.f; }
        for (int kp = 0; kp < nkt; kp += 2) {
          float p0[8], p1[8];
          {
            v16h bk = fragB_lds_masked(O1 + kp * 16 * AS + hoff, AS, lane, hd);
            v8f z = {}; v8f sc = wmma16(aq, bk, z);
#pragma unroll
            for (int r = 0; r < 8; r++) p0[r] = sc[r];
          }
          if (kp + 1 < nkt) {
            v16h bk = fragB_lds_masked(O1 + (kp + 1) * 16 * AS + hoff, AS, lane, hd);
            v8f z = {}; v8f sc = wmma16(aq, bk, z);
#pragma unroll
            for (int r = 0; r < 8; r++) p1[r] = sc[r];
          } else {
#pragma unroll
            for (int r = 0; r < 8; r++) p1[r] = -1e30f;
          }
          float nm[8], scl[8];
#pragma unroll
          for (int r = 0; r < 8; r++) {
            float tm = gmax16(fmaxf(p0[r], p1[r]));
            float nmr = fmaxf(om[r], tm);
            scl[r] = __expf(om[r] - nmr);
            om[r] = nmr; nm[r] = nmr;
          }
#pragma unroll
          for (int r = 0; r < 8; r++) {
            p0[r] = __expf(p0[r] - nm[r]);
            p1[r] = __expf(p1[r] - nm[r]);
            float rs = gsum16(p0[r] + p1[r]);
            os[r] = os[r] * scl[r] + rs;
            oacc[r] *= scl[r];
          }
#pragma unroll
          for (int r = 0; r < 8; r++) {        // P tile -> per-wave scratch (f16)
            int row = r + (g << 3);
            scr[row * 32 + ln]      = (half_t)p0[r];
            scr[row * 32 + 16 + ln] = (half_t)p1[r];
          }
          lds_fence();
          v16h ap = fragA_lds(scr, 32, lane);
          v16h bv = fragB_lds(Vt + hoff * VtS + kp * 16, VtS, lane);
          oacc = wmma16(ap, bv, oacc);
        }
#pragma unroll
        for (int r = 0; r < 8; r++) {
          if (ln < hd) {
            int row = qt * 16 + r + (g << 3);
            QO[row * AS + hoff + ln] = (half_t)(oacc[r] / os[r]);
          }
        }
      }
    }
  }
  __syncthreads();

  // ---- phase 3: dense + residual + LN1 -> O1 (overwrites K) ----
  {
    for (int rt = wave; rt < Srt; rt += NW) {
      v8f acc[8];
#pragma unroll
      for (int ct = 0; ct < 8; ct++) acc[ct] = (v8f){};
      for (int k0 = 0; k0 < A; k0 += 32) {
        v16h a = fragA_lds(QO + rt * 16 * AS + k0, AS, lane);
#pragma unroll
        for (int ct = 0; ct < 8; ct++) if (ct < nvt) {
          v16h b = fragB_gbl(P.wd, A, ct * 16, k0, lane);
          acc[ct] = wmma16(a, b, acc[ct]);
        }
      }
      float ps[8], sq[8];
#pragma unroll
      for (int r = 0; r < 8; r++) { ps[r] = 0.f; sq[r] = 0.f; }
#pragma unroll
      for (int ct = 0; ct < 8; ct++) if (ct < nvt) {
        float bc = P.bd[ct * 16 + ln];
#pragma unroll
        for (int r = 0; r < 8; r++) {
          int row = rt * 16 + r + (g << 3);
          float resid = (float)x_lds[row * XS + P.voff + ct * 16 + ln];
          float v = acc[ct][r] + bc + resid;
          acc[ct][r] = v; ps[r] += v; sq[r] += v * v;
        }
      }
      float mu[8], rstd[8];
#pragma unroll
      for (int r = 0; r < 8; r++) {
        float s = gsum16(ps[r]), s2 = gsum16(sq[r]);
        float m = s / (float)Dv;
        float var = s2 / (float)Dv - m * m;
        mu[r] = m; rstd[r] = rsqrtf(var + 1e-5f);
      }
#pragma unroll
      for (int ct = 0; ct < 8; ct++) if (ct < nvt) {
        float gc = P.g1[ct * 16 + ln], bb = P.bt1[ct * 16 + ln];
#pragma unroll
        for (int r = 0; r < 8; r++) {
          int row = rt * 16 + r + (g << 3);
          O1[row * AS + ct * 16 + ln] =
              (half_t)((acc[ct][r] - mu[r]) * rstd[r] * gc + bb);
        }
      }
    }
  }
  __syncthreads();

  // ---- phase 4: FFN (fc1+ReLU streamed through scratch, fc2) + LN2 -> global ----
  {
    for (int rt = wave; rt < Srt; rt += NW) {
      v8f facc[8];
#pragma unroll
      for (int ct = 0; ct < 8; ct++) facc[ct] = (v8f){};
      for (int jp = 0; jp < 32; jp += 2) {         // 512/16 hidden tiles, in pairs
#pragma unroll
        for (int t = 0; t < 2; t++) {
          int j0 = (jp + t) * 16;
          v8f h = {};
          for (int k0 = 0; k0 < Dv; k0 += 32) {
            v16h a = fragA_lds(O1 + rt * 16 * AS + k0, AS, lane);
            v16h b = fragB_gbl(P.w1, Dv, j0, k0, lane);
            h = wmma16(a, b, h);
          }
          float bc = P.b1[j0 + ln];
#pragma unroll
          for (int r = 0; r < 8; r++) {
            float v = fmaxf(h[r] + bc, 0.f);       // ReLU
            scr[(r + (g << 3)) * 32 + t * 16 + ln] = (half_t)v;
          }
        }
        lds_fence();
        v16h ah = fragA_lds(scr, 32, lane);
#pragma unroll
        for (int ct = 0; ct < 8; ct++) if (ct < nvt) {
          v16h b = fragB_gbl(P.w2, 512, ct * 16, jp * 16, lane);
          facc[ct] = wmma16(ah, b, facc[ct]);
        }
      }
      float ps[8], sq[8];
#pragma unroll
      for (int r = 0; r < 8; r++) { ps[r] = 0.f; sq[r] = 0.f; }
#pragma unroll
      for (int ct = 0; ct < 8; ct++) if (ct < nvt) {
        float bc = P.b2[ct * 16 + ln];
#pragma unroll
        for (int r = 0; r < 8; r++) {
          int row = rt * 16 + r + (g << 3);
          float o1v = (float)O1[row * AS + ct * 16 + ln];
          float v = facc[ct][r] + bc + o1v;
          facc[ct][r] = v; ps[r] += v; sq[r] += v * v;
        }
      }
      float mu[8], rstd[8];
#pragma unroll
      for (int r = 0; r < 8; r++) {
        float s = gsum16(ps[r]), s2 = gsum16(sq[r]);
        float m = s / (float)Dv;
        float var = s2 / (float)Dv - m * m;
        mu[r] = m; rstd[r] = rsqrtf(var + 1e-5f);
      }
#pragma unroll
      for (int ct = 0; ct < 8; ct++) if (ct < nvt) {
        float gc = P.g2[ct * 16 + ln], bb = P.bt2[ct * 16 + ln];
#pragma unroll
        for (int r = 0; r < 8; r++) {
          int row = rt * 16 + r + (g << 3);
          xg[base0 + (long)row * tstr + P.voff + ct * 16 + ln] =
              (facc[ct][r] - mu[r]) * rstd[r] * gc + bb;
        }
      }
    }
  }
}

// ===========================================================================
// Host side: convert weights to f16 once per call, then 8 fused block kernels.
// Input flattening assumption: d_in[0]=x, then per layer (insertion order
// xt,tx,xs,sx,tt,tt1,ss,ss1) 16 leaves: Wq.w,Wq.b,Wk.w,Wk.b,Wv.w,Wv.b,
// dense.w,dense.b,fc1.w,fc1.b,fc2.w,fc2.b,ln1.g,ln1.b,ln2.g,ln2.b.
// ===========================================================================
extern "C" void kernel_launch(void* const* d_in, const int* in_sizes, int n_in,
                              void* d_out, int out_size, void* d_ws, size_t ws_size,
                              hipStream_t stream) {
  (void)in_sizes; (void)n_in; (void)ws_size;
  float* xbuf = (float*)d_out;
  hipMemcpyAsync(xbuf, d_in[0], (size_t)out_size * sizeof(float),
                 hipMemcpyDeviceToDevice, stream);

  half_t* ws = (half_t*)d_ws;
  size_t off = 0;

  struct Cfg { int A, Dq, Dk, Dv, qoff, koff, voff, dim1; };
  const Cfg cfg[8] = {
      {64, 32, 64, 64, 96, 0, 0, 1},      // xt : q=T, k=v=O, dim 1
      {64, 64, 32, 32, 0, 96, 96, 1},     // tx : q=O, k=v=T, dim 1
      {64, 32, 64, 64, 64, 0, 0, 0},      // xs : q=S, k=v=O, dim 2
      {64, 64, 32, 32, 0, 64, 64, 0},     // sx : q=O, k=v=S, dim 2
      {128, 128, 128, 128, 0, 0, 0, 1},   // tt
      {128, 128, 128, 128, 0, 0, 0, 1},   // tt1
      {128, 128, 128, 128, 0, 0, 0, 0},   // ss
      {128, 128, 128, 128, 0, 0, 0, 0},   // ss1
  };

  for (int li = 0; li < 8; li++) {
    const Cfg& c = cfg[li];
    int base = 1 + 16 * li;
    int wsz[6]  = { c.A * c.Dq, c.A * c.Dk, c.A * c.Dv,
                    c.Dv * c.A, 512 * c.Dv, c.Dv * 512 };
    int widx[6] = { base + 0, base + 2, base + 4, base + 6, base + 8, base + 10 };
    const half_t* wp[6];
    for (int m = 0; m < 6; m++) {
      half_t* dst = ws + off;
      int n = wsz[m];
      cvt_f32_f16<<<(n + 255) / 256, 256, 0, stream>>>((const float*)d_in[widx[m]],
                                                       dst, n);
      wp[m] = dst;
      off += (size_t)n;
    }

    BlockParams P;
    P.x = xbuf;
    P.wq = wp[0]; P.wk = wp[1]; P.wv = wp[2];
    P.wd = wp[3]; P.w1 = wp[4]; P.w2 = wp[5];
    P.bq  = (const float*)d_in[base + 1];
    P.bk  = (const float*)d_in[base + 3];
    P.bv  = (const float*)d_in[base + 5];
    P.bd  = (const float*)d_in[base + 7];
    P.b1  = (const float*)d_in[base + 9];
    P.b2  = (const float*)d_in[base + 11];
    P.g1  = (const float*)d_in[base + 12];
    P.bt1 = (const float*)d_in[base + 13];
    P.g2  = (const float*)d_in[base + 14];
    P.bt2 = (const float*)d_in[base + 15];
    P.S = c.dim1 ? 48 : 256;
    P.A = c.A; P.hd = c.A / 8;
    P.Dq = c.Dq; P.Dk = c.Dk; P.Dv = c.Dv;
    P.qoff = c.qoff; P.koff = c.koff; P.voff = c.voff;
    P.dim1 = c.dim1;

    int nseq = c.dim1 ? (8 * 256) : (8 * 48);
    size_t shbytes =
        (size_t)(3 * P.S * 136 + (P.A + 16) * (P.S + 8) + NW * 512) * sizeof(half_t);
    msti_block_kernel<<<nseq, BLOCK, shbytes, stream>>>(P);
  }
}
